// InterlacedPoolAttention2_31748398252239
// MI455X (gfx1250) — matrix operand order repaired and checked by reading, not compiled, not run on hardware
//
#include <hip/hip_runtime.h>
#include <hip/hip_bf16.h>
#include <math.h>

typedef __attribute__((ext_vector_type(16))) __bf16 v16bf;
typedef __attribute__((ext_vector_type(8)))  float  v8f;

// Problem constants (from reference)
constexpr int Bn   = 8;
constexpr int Hh   = 56;
constexpr int Nn   = Hh * Hh;        // 3136
constexpr int Cc   = 384;
constexpr int CI   = 192;
constexpr int NHh  = 8;
constexpr int HD   = Cc / NHh;       // 48
constexpr int WS7  = 7;
constexpr int Ll   = WS7 * WS7;      // 49
constexpr int NbW  = 512;            // B * qh * qw = 8*8*8
constexpr int Mtot = Ll * NbW;       // 25088 tokens
constexpr int C3   = 3 * Cc;         // 1152
constexpr int KT   = 32;             // CAB keys per iteration
constexpr int NTIL = Nn / KT;        // 98

// ---------------------------------------------------------------------------
// WMMA fragment loaders (this code's contiguous k order, wave32)
// A 16x32 bf16: lane row = lane&15; kg = lane>>4; elems 0..15 -> K = kg*16 + e
// B 32x16 bf16: lane col = lane&15; elems 0..15 -> K = kg*16 + e
// C/D 16x16 f32: lane col = lane&15; vgpr r -> row = r + 8*kg
// ---------------------------------------------------------------------------
__device__ inline v16bf frag_a(const __bf16* A, int ld, int row, int k0, int kg) {
  v16bf a;
  const __bf16* p = A + (size_t)row * ld + k0 + kg * 16;
#pragma unroll
  for (int e = 0; e < 8; ++e) a[e] = p[e];
  p += 8;
#pragma unroll
  for (int e = 0; e < 8; ++e) a[8 + e] = p[e];
  return a;
}

// B[k][n] = W[n][k]  (W row-major N x K)  -> computes A @ W^T
__device__ inline v16bf frag_b_nt(const __bf16* W, int ld, int col, int k0, int kg) {
  v16bf b;
  const __bf16* p = W + (size_t)col * ld + k0 + kg * 16;
#pragma unroll
  for (int e = 0; e < 16; ++e) b[e] = p[e];
  return b;
}

// B[k][n] = Bm[k0+k][colbase+n]  (Bm row-major K x ld) -> computes A @ B
__device__ inline v16bf frag_b_nn(const __bf16* Bm, int ld, int colbase, int k0,
                                  int kg, int ln) {
  v16bf b;
#pragma unroll
  for (int e = 0; e < 16; ++e)
    b[e] = Bm[(size_t)(k0 + kg * 16 + e) * ld + colbase + ln];
  return b;
}

__device__ inline v8f wmma_bf16(v16bf a, v16bf b, v8f c) {
  return __builtin_amdgcn_wmma_f32_16x16x32_bf16(false, a, false, b, (short)0, c,
                                                 false, false);
}

// ---------------------------------------------------------------------------
// Async Global->LDS staging (gfx1250 ASYNCcnt path, CDNA5 ISA)
// ---------------------------------------------------------------------------
__device__ inline unsigned lds_off(const void* p) {
  // generic pointer to LDS: addr[31:0] is the LDS byte offset
  return (unsigned)(unsigned long long)p;
}

// Copy one contiguous 12KB tile (KT*CI bf16) to LDS: 768 x 16B chunks,
// 128 threads -> 6 wave-wide async instructions per wave (ASYNCcnt += 6).
__device__ inline void async_copy_tile(const __bf16* gsrc, unsigned lds_base,
                                       int tid) {
#pragma unroll
  for (int i = 0; i < 6; ++i) {
    const int chunk = tid + i * 128;
    unsigned laddr = lds_base + chunk * 16;
    const __bf16* g = gsrc + chunk * 8;
    asm volatile("global_load_async_to_lds_b128 %0, %1, off"
                 :: "v"(laddr), "v"(g)
                 : "memory");
  }
}

// ---------------------------------------------------------------------------
// fp32 -> bf16 conversion
// ---------------------------------------------------------------------------
__global__ void cvt_bf16(const float* __restrict__ src, __bf16* __restrict__ dst,
                         size_t n) {
  size_t i = (size_t)blockIdx.x * blockDim.x + threadIdx.x;
  if (i < n) dst[i] = (__bf16)src[i];
}

// ---------------------------------------------------------------------------
// Generic bf16 GEMM: out = A(MxK) @ W(NxK)^T, f32 accumulate.
// 128 threads = 4 waves, each wave one 16x16 tile, block covers 32x32.
// mode 0: outb = bf16(alpha * acc)
// mode 1: outb = bf16((acc + bias[col]) * (col<qcols ? qscale : 1))   (qkv)
// mode 2: outb = bf16(acc + resid[row*N+col])                          (CAB out)
// mode 3: outf[window-reverse-permute(row), col] = acc + bias[col]     (final)
// ---------------------------------------------------------------------------
__global__ __launch_bounds__(128) void gemm_wt(
    const __bf16* __restrict__ A, const __bf16* __restrict__ W, int M, int N,
    int K, int mode, float alpha, const float* __restrict__ bias, float qscale,
    int qcols, const float* __restrict__ resid, __bf16* __restrict__ outb,
    float* __restrict__ outf) {
  const int lane = threadIdx.x & 31;
  const int wave = threadIdx.x >> 5;
  const int tileN = (blockIdx.x * 2 + (wave & 1)) * 16;
  const int tileM = (blockIdx.y * 2 + (wave >> 1)) * 16;
  if (tileM >= M || tileN >= N) return;
  const int kg = lane >> 4, ln = lane & 15;

  v8f acc = {};
  for (int k0 = 0; k0 < K; k0 += 32) {
    v16bf a = frag_a(A, K, tileM + ln, k0, kg);
    v16bf b = frag_b_nt(W, K, tileN + ln, k0, kg);
    acc = wmma_bf16(a, b, acc);
  }

  const int col = tileN + ln;
#pragma unroll
  for (int r = 0; r < 8; ++r) {
    const int row = tileM + r + 8 * kg;
    float v = acc[r];
    if (mode == 0) {
      v *= alpha;
    } else if (mode == 1) {
      v = (v + bias[col]) * (col < qcols ? qscale : 1.0f);
    } else if (mode == 2) {
      v = v + resid[(size_t)row * N + col];
    } else {  // mode 3: reverse window permute into (B, H, W, C) fp32 output
      v = v + bias[col];
      const int l = row / NbW, nb = row % NbW;
      const int ph = l / WS7, pw = l % WS7;
      const int bb = nb >> 6, ii = (nb >> 3) & 7, jj = nb & 7;
      const int n_out = (ii * WS7 + ph) * Hh + (jj * WS7 + pw);
      outf[((size_t)bb * Nn + n_out) * Cc + col] = v;
      continue;
    }
    outb[(size_t)row * N + col] = (__bf16)v;
  }
}

// ---------------------------------------------------------------------------
// CAB flash attention: out = softmax(t p^T) g  per batch, N=3136, CI=192.
// 4 waves/block = 64 query rows; 32-key tiles double-buffered in LDS via
// global_load_async_to_lds_b128 (prefetch jt+1 while computing jt).
// ---------------------------------------------------------------------------
__global__ __launch_bounds__(128) void cab_attn(const __bf16* __restrict__ t,
                                                const __bf16* __restrict__ p,
                                                const __bf16* __restrict__ g,
                                                __bf16* __restrict__ out) {
  __shared__ __bf16 Pk[2][KT * CI];   // key tiles   (12 KB each)
  __shared__ __bf16 Gv[2][KT * CI];   // value tiles (12 KB each)
  __shared__ __bf16 Ps[4][16 * KT];   // per-wave P staging (1 KB each)

  const int tid = threadIdx.x;
  const int wave = tid >> 5, lane = tid & 31;
  const int kg = lane >> 4, ln = lane & 15;
  const int b = blockIdx.y;
  const int rowTile = blockIdx.x * 64 + wave * 16;
  const __bf16* tb = t + (size_t)b * Nn * CI;
  const __bf16* pb = p + (size_t)b * Nn * CI;
  const __bf16* gb = g + (size_t)b * Nn * CI;

  v16bf at[6];
#pragma unroll
  for (int kk = 0; kk < 6; ++kk)
    at[kk] = frag_a(tb, CI, rowTile + ln, kk * 32, kg);

  const v8f vz = {};
  v8f O[12];
#pragma unroll
  for (int i = 0; i < 12; ++i) O[i] = vz;
  float m8[8], l8[8];
#pragma unroll
  for (int r = 0; r < 8; ++r) { m8[r] = -3.0e38f; l8[r] = 0.f; }

  // prologue: stage tile 0 (12 async instructions per wave)
  async_copy_tile(pb, lds_off(&Pk[0][0]), tid);
  async_copy_tile(gb, lds_off(&Gv[0][0]), tid);

  for (int jt = 0; jt < NTIL; ++jt) {
    const int cur = jt & 1;
    if (jt + 1 < NTIL) {
      // prefetch next tile into the other buffer (12 more in flight)
      async_copy_tile(pb + (size_t)(jt + 1) * KT * CI, lds_off(&Pk[1 - cur][0]), tid);
      async_copy_tile(gb + (size_t)(jt + 1) * KT * CI, lds_off(&Gv[1 - cur][0]), tid);
      asm volatile("s_wait_asynccnt 0xc" ::: "memory");  // current buffer done
    } else {
      asm volatile("s_wait_asynccnt 0x0" ::: "memory");
    }
    __syncthreads();  // cross-wave visibility of staged tiles

    const __bf16* kt_ = &Pk[cur][0];
    const __bf16* gt_ = &Gv[cur][0];

    v8f S0 = vz, S1 = vz;
#pragma unroll
    for (int kk = 0; kk < 6; ++kk)
      S0 = wmma_bf16(at[kk], frag_b_nt(kt_, CI, ln, kk * 32, kg), S0);
#pragma unroll
    for (int kk = 0; kk < 6; ++kk)
      S1 = wmma_bf16(at[kk], frag_b_nt(kt_, CI, 16 + ln, kk * 32, kg), S1);

    // online softmax row update (row = r + 8*kg; 16 cols per half in lanes)
#pragma unroll
    for (int r = 0; r < 8; ++r) {
      float mx = fmaxf(S0[r], S1[r]);
      for (int off = 1; off < 16; off <<= 1) mx = fmaxf(mx, __shfl_xor(mx, off, 32));
      const float mnew = fmaxf(m8[r], mx);
      const float sc = __expf(m8[r] - mnew);
      const float p0 = __expf(S0[r] - mnew);
      const float p1 = __expf(S1[r] - mnew);
      float rs = p0 + p1;
      for (int off = 1; off < 16; off <<= 1) rs += __shfl_xor(rs, off, 32);
      l8[r] = l8[r] * sc + rs;
      m8[r] = mnew;
#pragma unroll
      for (int cb = 0; cb < 12; ++cb) O[cb][r] *= sc;
      const int prow = r + 8 * kg;
      Ps[wave][prow * KT + ln] = (__bf16)p0;
      Ps[wave][prow * KT + 16 + ln] = (__bf16)p1;
    }
    // same-wave LDS RAW is kept in order by hardware; no barrier needed here
    v16bf pa = frag_a(&Ps[wave][0], KT, ln, 0, kg);
#pragma unroll
    for (int cb = 0; cb < 12; ++cb)
      O[cb] = wmma_bf16(pa, frag_b_nn(gt_, CI, cb * 16, 0, kg, ln), O[cb]);

    __syncthreads();  // all waves done with this buffer before it is re-staged
  }

  __bf16* ob = out + (size_t)b * Nn * CI;
#pragma unroll
  for (int cb = 0; cb < 12; ++cb)
#pragma unroll
    for (int r = 0; r < 8; ++r) {
      const int row = rowTile + r + 8 * kg;
      ob[(size_t)row * CI + cb * 16 + ln] = (__bf16)(O[cb][r] / l8[r]);
    }
}

// ---------------------------------------------------------------------------
// Window permute: x4 (B,H,W,C) -> xp ((ph pw),(b i j), C), both bf16.
// xp[l, nb, c] = x4[b, j*7+pw, i*7+ph, c]   (includes the (0,2,1,3) transpose)
// ---------------------------------------------------------------------------
__global__ void window_permute(const __bf16* __restrict__ x4,
                               __bf16* __restrict__ xp) {
  size_t idx = (size_t)blockIdx.x * blockDim.x + threadIdx.x;
  const size_t total = (size_t)Mtot * Cc;
  if (idx >= total) return;
  const int c = (int)(idx % Cc);
  const int row = (int)(idx / Cc);
  const int nb = row % NbW, l = row / NbW;
  const int ph = l / WS7, pw = l % WS7;
  const int bb = nb >> 6, ii = (nb >> 3) & 7, jj = nb & 7;
  const int n_src = (jj * WS7 + pw) * Hh + (ii * WS7 + ph);
  xp[idx] = x4[((size_t)bb * Nn + n_src) * Cc + c];
}

// ---------------------------------------------------------------------------
// Windowed MHA: per (head, window) block; 49x49 scores + rel-pos bias,
// softmax, P@V. Q/K/V zero-padded to 64x64 in LDS; 4 waves x 16-row strips.
// q already scaled by hd^-0.5 in the qkv GEMM epilogue.
// ---------------------------------------------------------------------------
__global__ __launch_bounds__(128) void win_mha(const __bf16* __restrict__ qkv,
                                               const float* __restrict__ rpb,
                                               const int* __restrict__ relidx,
                                               __bf16* __restrict__ o) {
  __shared__ __bf16 Qs[64 * 64], Ks[64 * 64], Vs[64 * 64];
  __shared__ __bf16 Ps[4][16 * 64];
  __shared__ float Bs[Ll * Ll];
  const int h = blockIdx.x;
  const int nb = blockIdx.y;
  const int tid = threadIdx.x;

  for (int idx = tid; idx < 64 * 64; idx += 128) {
    const int l = idx >> 6, d = idx & 63;
    __bf16 qv = (__bf16)0.f, kv = (__bf16)0.f, vv = (__bf16)0.f;
    if (l < Ll && d < HD) {
      const size_t ro = ((size_t)l * NbW + nb) * C3;
      qv = qkv[ro + h * HD + d];
      kv = qkv[ro + Cc + h * HD + d];
      vv = qkv[ro + 2 * Cc + h * HD + d];
    }
    Qs[idx] = qv; Ks[idx] = kv; Vs[idx] = vv;
  }
  for (int idx = tid; idx < Ll * Ll; idx += 128)
    Bs[idx] = rpb[relidx[idx] * NHh + h];
  __syncthreads();

  const int wave = tid >> 5, lane = tid & 31;
  const int kg = lane >> 4, ln = lane & 15;
  const int rt = wave * 16;
  const v8f vz = {};

  v8f S[4];
#pragma unroll
  for (int ct = 0; ct < 4; ++ct) {
    v8f s = vz;
#pragma unroll
    for (int kt = 0; kt < 2; ++kt) {
      v16bf a = frag_a(Qs, 64, rt + ln, kt * 32, kg);
      v16bf b = frag_b_nt(Ks, 64, ct * 16 + ln, kt * 32, kg);
      s = wmma_bf16(a, b, s);
    }
    const int col = ct * 16 + ln;
#pragma unroll
    for (int r = 0; r < 8; ++r) {
      const int row = rt + r + 8 * kg;
      float v = s[r];
      if (row < Ll && col < Ll) v += Bs[row * Ll + col];
      else v = -1.0e30f;  // mask padded key columns
      s[r] = v;
    }
    S[ct] = s;
  }

  float l8[8];
#pragma unroll
  for (int r = 0; r < 8; ++r) {
    float mx = -3.0e38f;
#pragma unroll
    for (int ct = 0; ct < 4; ++ct) mx = fmaxf(mx, S[ct][r]);
    for (int off = 1; off < 16; off <<= 1) mx = fmaxf(mx, __shfl_xor(mx, off, 32));
    float rs = 0.f;
    const int prow = (r + 8 * kg) * 64;
#pragma unroll
    for (int ct = 0; ct < 4; ++ct) {
      const float pv = __expf(S[ct][r] - mx);
      Ps[wave][prow + ct * 16 + ln] = (__bf16)pv;
      rs += pv;
    }
    for (int off = 1; off < 16; off <<= 1) rs += __shfl_xor(rs, off, 32);
    l8[r] = rs;
  }
  // same-wave LDS RAW: hardware keeps LDS in order per wave

#pragma unroll
  for (int cb = 0; cb < 3; ++cb) {
    v8f Ov = vz;
#pragma unroll
    for (int kt = 0; kt < 2; ++kt) {
      v16bf a = frag_a(&Ps[wave][0], 64, ln, kt * 32, kg);
      v16bf b = frag_b_nn(Vs, 64, cb * 16, kt * 32, kg, ln);
      Ov = wmma_bf16(a, b, Ov);
    }
#pragma unroll
    for (int r = 0; r < 8; ++r) {
      const int row = rt + r + 8 * kg;
      if (row < Ll)
        o[((size_t)row * NbW + nb) * Cc + h * HD + cb * 16 + ln] =
            (__bf16)(Ov[r] / l8[r]);
    }
  }
}

// ---------------------------------------------------------------------------
extern "C" void kernel_launch(void* const* d_in, const int* in_sizes, int n_in,
                              void* d_out, int out_size, void* d_ws,
                              size_t ws_size, hipStream_t stream) {
  (void)in_sizes; (void)n_in; (void)out_size; (void)ws_size;
  const float* x          = (const float*)d_in[0];
  const float* theta_w    = (const float*)d_in[4];
  const float* phi_w      = (const float*)d_in[5];
  const float* g_w        = (const float*)d_in[6];
  const float* cab_out_w  = (const float*)d_in[7];
  const float* in_proj_w  = (const float*)d_in[8];
  const float* in_proj_b  = (const float*)d_in[9];
  const float* out_proj_w = (const float*)d_in[10];
  const float* out_proj_b = (const float*)d_in[11];
  const float* rpb        = (const float*)d_in[12];
  const int*   relidx     = (const int*)d_in[13];
  float* out = (float*)d_out;

  char* ws = (char*)d_ws;
  size_t off = 0;
  auto alloc = [&](size_t bytes) -> void* {
    void* p = ws + off;
    off += (bytes + 255) & ~(size_t)255;
    return p;
  };
  // Aliased layout (~118 MB total):
  __bf16* x_bf   = (__bf16*)alloc((size_t)Mtot * Cc * 2);  // later reused as x4
  __bf16* t_bf   = (__bf16*)alloc((size_t)Mtot * CI * 2);  // t|p reused as xp
  __bf16* p_bf   = (__bf16*)alloc((size_t)Mtot * CI * 2);
  __bf16* g_bf   = (__bf16*)alloc((size_t)Mtot * CI * 2);  // g|attn reused as o
  __bf16* a_bf   = (__bf16*)alloc((size_t)Mtot * CI * 2);
  __bf16* qkv_bf = (__bf16*)alloc((size_t)Mtot * C3 * 2);
  __bf16* wth_bf = (__bf16*)alloc((size_t)CI * Cc * 2);
  __bf16* wph_bf = (__bf16*)alloc((size_t)CI * Cc * 2);
  __bf16* wg_bf  = (__bf16*)alloc((size_t)CI * Cc * 2);
  __bf16* wo_bf  = (__bf16*)alloc((size_t)Cc * CI * 2);
  __bf16* wip_bf = (__bf16*)alloc((size_t)C3 * Cc * 2);
  __bf16* wop_bf = (__bf16*)alloc((size_t)Cc * Cc * 2);
  __bf16* x4_bf = x_bf;
  __bf16* xp_bf = t_bf;  // t+p contiguous: Mtot*Cc bf16
  __bf16* o_bf  = g_bf;  // g+attn contiguous: Mtot*Cc bf16

  auto cvt = [&](const float* s, __bf16* d, size_t n) {
    cvt_bf16<<<(unsigned)((n + 255) / 256), 256, 0, stream>>>(s, d, n);
  };
  cvt(x, x_bf, (size_t)Mtot * Cc);
  cvt(theta_w, wth_bf, (size_t)CI * Cc);
  cvt(phi_w, wph_bf, (size_t)CI * Cc);
  cvt(g_w, wg_bf, (size_t)CI * Cc);
  cvt(cab_out_w, wo_bf, (size_t)Cc * CI);
  cvt(in_proj_w, wip_bf, (size_t)C3 * Cc);
  cvt(out_proj_w, wop_bf, (size_t)Cc * Cc);

  const float cab_scale = 1.0f / sqrtf((float)CI);
  const float q_scale = 1.0f / sqrtf((float)HD);
  dim3 blk(128);

  // t/p/g projections: (Mtot x 384) @ (192 x 384)^T
  dim3 gP(CI / 32, Mtot / 32);
  gemm_wt<<<gP, blk, 0, stream>>>(x_bf, wth_bf, Mtot, CI, Cc, 0, cab_scale,
                                  nullptr, 0.f, 0, nullptr, t_bf, nullptr);
  gemm_wt<<<gP, blk, 0, stream>>>(x_bf, wph_bf, Mtot, CI, Cc, 0, 1.0f, nullptr,
                                  0.f, 0, nullptr, p_bf, nullptr);
  gemm_wt<<<gP, blk, 0, stream>>>(x_bf, wg_bf, Mtot, CI, Cc, 0, 1.0f, nullptr,
                                  0.f, 0, nullptr, g_bf, nullptr);

  // CAB flash attention: 4 waves/block, async double-buffered K/V tiles
  cab_attn<<<dim3(Nn / 64, Bn), dim3(128), 0, stream>>>(t_bf, p_bf, g_bf, a_bf);

  // x4 = x + attn @ cab_out_w^T   (K=192, N=384), bf16 out (overwrites x_bf)
  gemm_wt<<<dim3(Cc / 32, Mtot / 32), blk, 0, stream>>>(
      a_bf, wo_bf, Mtot, Cc, CI, 2, 1.0f, nullptr, 0.f, 0, x, x4_bf, nullptr);

  // window permute into xp
  {
    size_t n = (size_t)Mtot * Cc;
    window_permute<<<(unsigned)((n + 255) / 256), 256, 0, stream>>>(x4_bf, xp_bf);
  }

  // qkv = xp @ in_proj_w^T + b ; q scaled by hd^-0.5
  gemm_wt<<<dim3(C3 / 32, Mtot / 32), blk, 0, stream>>>(
      xp_bf, wip_bf, Mtot, C3, Cc, 1, 1.0f, in_proj_b, q_scale, Cc, nullptr,
      qkv_bf, nullptr);

  // windowed attention
  win_mha<<<dim3(NHh, NbW), dim3(128), 0, stream>>>(qkv_bf, rpb, relidx, o_bf);

  // final: out = reverse_permute(o @ out_proj_w^T + b), fp32
  gemm_wt<<<dim3(Cc / 32, Mtot / 32), blk, 0, stream>>>(
      o_bf, wop_bf, Mtot, Cc, Cc, 3, 1.0f, out_proj_b, 0.f, 0, nullptr, nullptr,
      out);
}
